// kNN_41772851921312
// MI455X (gfx1250) — matrix operand order repaired and checked by reading, not compiled, not run on hardware
//
#include <hip/hip_runtime.h>

typedef __attribute__((ext_vector_type(16))) _Float16 v16h;
typedef __attribute__((ext_vector_type(8)))  _Float16 v8h;
typedef __attribute__((ext_vector_type(8)))  float    v8f;

#define D_DIM    512
#define ROWS     32           // rows per workgroup (two 16-row WMMA tiles)
#define COLS     128          // columns per strip (8 compute waves x 16)
#define KCHUNKS  16           // 512 / 32
#define KSEL     16           // top-k
#define NTHREADS 320          // 8 compute waves + 2 selection waves

// ---------------------------------------------------------------------------
// Kernel 0: per-row squared norms, one wave32 per row (coalesced).
// ---------------------------------------------------------------------------
__global__ void sqnorm_kernel(const float* __restrict__ x, float* __restrict__ sq, int N) {
  int wv   = (int)((blockIdx.x * blockDim.x + threadIdx.x) >> 5);
  int lane = (int)(threadIdx.x & 31);
  if (wv >= N) return;
  const float* row = x + (size_t)wv * D_DIM;
  float s = 0.0f;
  #pragma unroll
  for (int i = 0; i < D_DIM / 32; ++i) {
    float v = row[lane + 32 * i];
    s += v * v;
  }
  #pragma unroll
  for (int m = 16; m > 0; m >>= 1) s += __shfl_xor(s, m, 32);
  if (lane == 0) sq[wv] = s;
}

// ---------------------------------------------------------------------------
// Kernel 0b: one-time fp32 -> f16 conversion of x (B-operand source).
// ---------------------------------------------------------------------------
__global__ void cvt_kernel(const float* __restrict__ x, _Float16* __restrict__ xh, int total) {
  int i = (int)(blockIdx.x * blockDim.x + threadIdx.x) * 8;
  if (i >= total) return;
  float4 f0 = *(const float4*)(x + i);
  float4 f1 = *(const float4*)(x + i + 4);
  v8h h;
  h[0] = (_Float16)f0.x; h[1] = (_Float16)f0.y; h[2] = (_Float16)f0.z; h[3] = (_Float16)f0.w;
  h[4] = (_Float16)f1.x; h[5] = (_Float16)f1.y; h[6] = (_Float16)f1.z; h[7] = (_Float16)f1.w;
  *(v8h*)(xh + i) = h;
}

// Pack 16 consecutive fp32 into a v16h WMMA operand slice (fallback path).
__device__ __forceinline__ v16h pack16(const float* __restrict__ p) {
  float4 f0 = *(const float4*)(p + 0);
  float4 f1 = *(const float4*)(p + 4);
  float4 f2 = *(const float4*)(p + 8);
  float4 f3 = *(const float4*)(p + 12);
  v16h r;
  r[0]  = (_Float16)f0.x; r[1]  = (_Float16)f0.y; r[2]  = (_Float16)f0.z; r[3]  = (_Float16)f0.w;
  r[4]  = (_Float16)f1.x; r[5]  = (_Float16)f1.y; r[6]  = (_Float16)f1.z; r[7]  = (_Float16)f1.w;
  r[8]  = (_Float16)f2.x; r[9]  = (_Float16)f2.y; r[10] = (_Float16)f2.z; r[11] = (_Float16)f2.w;
  r[12] = (_Float16)f3.x; r[13] = (_Float16)f3.y; r[14] = (_Float16)f3.z; r[15] = (_Float16)f3.w;
  return r;
}

// ---------------------------------------------------------------------------
// Kernel 1: fused WMMA Gram tiles + overlapped streaming top-16 + averaging.
// Grid: N/32 blocks of 320 threads. Waves 0-7 compute, waves 8-9 select.
// ---------------------------------------------------------------------------
template <bool F16B>
__global__ void __launch_bounds__(NTHREADS) knn_kernel(const float* __restrict__ x,
                                                       const _Float16* __restrict__ xh,
                                                       const float* __restrict__ sq,
                                                       float* __restrict__ out, int N) {
  __shared__ _Float16 ldsA[2][16][D_DIM];      // 32 KB: two 16-row A tiles (f16)
  __shared__ float    lds_key[2][ROWS][COLS];  // 32 KB: double-buffered keys
  __shared__ float    pd[2][ROWS][KSEL];       // partial top-16 distances
  __shared__ int      pi[2][ROWS][KSEL];       // partial top-16 indices
  __shared__ int      lds_idx[ROWS][KSEL];     // merged neighbor indices

  const int tid     = (int)threadIdx.x;
  const int lane    = tid & 31;
  const int wave    = tid >> 5;
  const int rowBase = (int)blockIdx.x * ROWS;

  // --- Stage A tiles (32 rows x 512) into LDS as f16 ---
  for (int e = tid; e < ROWS * D_DIM; e += NTHREADS) {
    int r = e >> 9, d = e & (D_DIM - 1);
    ldsA[r >> 4][r & 15][d] = (_Float16)x[(size_t)(rowBase + r) * D_DIM + d];
  }
  __syncthreads();

  // WMMA operand layout constants (16-bit, 16x16x32):
  const int mA    = lane & 15;
  const int aOfs  = (lane >> 4) * 16;  // A: lanes>=16 hold K 16..31
  const int nB    = lane & 15;
  const int bOfs  = (lane >> 4) * 16;  // B: lanes>=16 hold K 16..31
  const int cN    = lane & 15;
  const int cMofs = (lane >> 4) * 8;   // C: lanes>=16 hold M g+8

  // Selection lane mapping (waves 8-9: 64 lanes, 2 per row, 64 cols each)
  const int selLane = tid - 256;
  const int selRow  = selLane & 31;
  const int selHalf = (selLane >> 5) & 1;
  const int selGrow = rowBase + selRow;

  float bd[KSEL];
  int   bi[KSEL];
  #pragma unroll
  for (int j = 0; j < KSEL; ++j) { bd[j] = 3.0e38f; bi[j] = 0; }

  const int NITER = N / COLS;
  for (int it = 0; it < NITER; ++it) {
    const int c0  = it * COLS;
    const int buf = it & 1;

    if (wave < 8) {
      const int colB = c0 + wave * 16 + nB;
      v8f acc0 = {}, acc1 = {};
      if constexpr (F16B) {
        const _Float16* brow = xh + (size_t)colB * D_DIM;
        #pragma unroll
        for (int kk = 0; kk < KCHUNKS; ++kk) {
          v16h b = *(const v16h*)(brow + kk * 32 + bOfs);   // 32B contiguous f16
          const int k0 = kk * 32 + aOfs;
          v8h a0lo = *(const v8h*)&ldsA[0][mA][k0];
          v8h a0hi = *(const v8h*)&ldsA[0][mA][k0 + 8];
          v8h a1lo = *(const v8h*)&ldsA[1][mA][k0];
          v8h a1hi = *(const v8h*)&ldsA[1][mA][k0 + 8];
          v16h a0 = __builtin_shufflevector(a0lo, a0hi, 0,1,2,3,4,5,6,7,8,9,10,11,12,13,14,15);
          v16h a1 = __builtin_shufflevector(a1lo, a1hi, 0,1,2,3,4,5,6,7,8,9,10,11,12,13,14,15);
          acc0 = __builtin_amdgcn_wmma_f32_16x16x32_f16(false, a0, false, b, (short)0, acc0, false, false);
          acc1 = __builtin_amdgcn_wmma_f32_16x16x32_f16(false, a1, false, b, (short)0, acc1, false, false);
        }
      } else {
        const float* brow = x + (size_t)colB * D_DIM;
        #pragma unroll
        for (int kk = 0; kk < KCHUNKS; ++kk) {
          v16h b = pack16(brow + kk * 32 + bOfs);
          const int k0 = kk * 32 + aOfs;
          v8h a0lo = *(const v8h*)&ldsA[0][mA][k0];
          v8h a0hi = *(const v8h*)&ldsA[0][mA][k0 + 8];
          v8h a1lo = *(const v8h*)&ldsA[1][mA][k0];
          v8h a1hi = *(const v8h*)&ldsA[1][mA][k0 + 8];
          v16h a0 = __builtin_shufflevector(a0lo, a0hi, 0,1,2,3,4,5,6,7,8,9,10,11,12,13,14,15);
          v16h a1 = __builtin_shufflevector(a1lo, a1hi, 0,1,2,3,4,5,6,7,8,9,10,11,12,13,14,15);
          acc0 = __builtin_amdgcn_wmma_f32_16x16x32_f16(false, a0, false, b, (short)0, acc0, false, false);
          acc1 = __builtin_amdgcn_wmma_f32_16x16x32_f16(false, a1, false, b, (short)0, acc1, false, false);
        }
      }
      // Selection key: sq[c] - 2*G[r][c] (row-constant sq[r] cannot change rank)
      const float sqc = sq[c0 + wave * 16 + cN];
      #pragma unroll
      for (int g = 0; g < 8; ++g) {
        lds_key[buf][cMofs + g][wave * 16 + cN]      = sqc - 2.0f * acc0[g];
        lds_key[buf][16 + cMofs + g][wave * 16 + cN] = sqc - 2.0f * acc1[g];
      }
    }
    __syncthreads();   // keys(it) visible; also fences reuse of buf from it-2

    if (wave >= 8) {   // overlaps with compute of strip it+1 on waves 0-7
      const int jb = selHalf * 64;
      for (int j = 0; j < 64; ++j) {
        float v  = lds_key[buf][selRow][jb + j];
        int   ci = c0 + jb + j;
        if (ci != selGrow && v < bd[KSEL - 1]) {
          bd[KSEL - 1] = v; bi[KSEL - 1] = ci;
          #pragma unroll
          for (int t = KSEL - 1; t > 0; --t) {
            if (bd[t] < bd[t - 1]) {
              float tv = bd[t]; bd[t] = bd[t - 1]; bd[t - 1] = tv;
              int   ti = bi[t]; bi[t] = bi[t - 1]; bi[t - 1] = ti;
            }
          }
        }
      }
    }
  }
  __syncthreads();

  // --- Publish partial lists, merge two sorted 16-lists per row ---
  if (wave >= 8) {
    #pragma unroll
    for (int j = 0; j < KSEL; ++j) {
      pd[selHalf][selRow][j] = bd[j];
      pi[selHalf][selRow][j] = bi[j];
    }
  }
  __syncthreads();

  if (tid < ROWS) {
    int p0 = 0, p1 = 0;
    for (int j = 0; j < KSEL; ++j) {
      float v0 = pd[0][tid][p0], v1 = pd[1][tid][p1];
      if (v0 <= v1) { lds_idx[tid][j] = pi[0][tid][p0]; ++p0; }
      else          { lds_idx[tid][j] = pi[1][tid][p1]; ++p1; }
    }
  }
  __syncthreads();

  // --- Cooperative gather-average of the 16 neighbor rows ---
  for (int e = tid; e < ROWS * D_DIM; e += NTHREADS) {
    int r = e >> 9, d = e & (D_DIM - 1);
    float s = 0.0f;
    #pragma unroll
    for (int j = 0; j < KSEL; ++j) {
      s += x[(size_t)lds_idx[r][j] * D_DIM + d];
    }
    out[(size_t)(rowBase + r) * D_DIM + d] = s * (1.0f / 16.0f);
  }
}

// ---------------------------------------------------------------------------
extern "C" void kernel_launch(void* const* d_in, const int* in_sizes, int n_in,
                              void* d_out, int out_size, void* d_ws, size_t ws_size,
                              hipStream_t stream) {
  (void)n_in; (void)out_size;
  const float* x   = (const float*)d_in[0];   // [N, 512] fp32
  float*       out = (float*)d_out;           // [N, 512] fp32

  const int D = D_DIM;
  const int N = in_sizes[0] / D;              // 8192

  // Workspace layout: sq[N] fp32, then (optionally) xh[N*D] f16.
  size_t sqBytes = (size_t)N * sizeof(float);
  size_t xhOff   = (sqBytes + 255) & ~(size_t)255;
  size_t need    = xhOff + (size_t)N * D * sizeof(_Float16);
  float*     sq = (float*)d_ws;
  _Float16*  xh = (_Float16*)((char*)d_ws + xhOff);
  const bool f16path = (ws_size >= need);

  // 1) per-row squared norms (one wave per row)
  {
    int blocks = (N * 32 + 255) / 256;
    sqnorm_kernel<<<blocks, 256, 0, stream>>>(x, sq, N);
  }
  // 1b) one-time f16 conversion of x (if workspace allows)
  if (f16path) {
    int total  = N * D;
    int blocks = (total / 8 + 255) / 256;
    cvt_kernel<<<blocks, 256, 0, stream>>>(x, xh, total);
  }
  // 2) fused WMMA distance tiles + overlapped top-16 + averaging
  if (f16path) {
    knn_kernel<true><<<N / ROWS, NTHREADS, 0, stream>>>(x, xh, sq, out, N);
  } else {
    knn_kernel<false><<<N / ROWS, NTHREADS, 0, stream>>>(x, nullptr, sq, out, N);
  }
}
